// self_BiLSTM_sparse_attn_predict_55319178772802
// MI455X (gfx1250) — matrix operand order repaired and checked by reading, not compiled, not run on hardware
//
#include <hip/hip_runtime.h>

// ---------------- problem constants ----------------
constexpr int kB    = 64;
constexpr int kS    = 512;
constexpr int kH    = 256;
constexpr int kH2   = 512;
constexpr int kTMAX = kS + 1;      // 513
constexpr int kTOPK = 5;
constexpr float kEPS = 1e-7f;
constexpr float kNEG = -1000000000.0f;

// ---------------- launch shape ----------------
constexpr int kNWG = 16;           // 16 workgroups, each owns 32 cols of h_comb
constexpr int kTPB = 256;          // 8 wave32 per workgroup

// ---------------- LDS layout (dynamic) ----------------
constexpr int WST = 264;           // bf16 row stride for weight/h tiles (pad vs banks)
constexpr int GST = 68;            // f32 row stride for gate tile
constexpr size_t OFF_W   = 0;                          // 128 x 264 bf16 = 67584 B
constexpr size_t OFF_HL  = OFF_W  + 128 * WST * 2;     // 64 x 264 bf16  = 33792 B
constexpr size_t OFF_G   = OFF_HL + 64 * WST * 2;      // 128 x 68 f32   = 34816 B
constexpr size_t OFF_C   = OFF_G  + 128 * GST * 4;     // 32 x 64 f32    = 8192 B
constexpr size_t OFF_HS  = OFF_C  + 32 * 64 * 4;       // 32 x 64 f32    = 8192 B
constexpr size_t OFF_TV  = OFF_HS + 32 * 64 * 4;       // 5 x 64 f32
constexpr size_t OFF_TI  = OFF_TV + 5 * 64 * 4;        // 5 x 64 i32
constexpr size_t OFF_AV  = OFF_TI + 5 * 64 * 4;        // 5 x 64 f32
constexpr size_t OFF_AI  = OFF_AV + 5 * 64 * 4;        // 5 x 64 i32
constexpr size_t OFF_XV  = OFF_AI + 5 * 64 * 4;        // 64 f32
constexpr size_t OFF_WIH = OFF_XV + 64 * 4;            // 128 f32
constexpr size_t OFF_BIA = OFF_WIH + 128 * 4;          // 128 f32
constexpr size_t OFF_W1  = OFF_BIA + 128 * 4;          // 32 f32
constexpr size_t OFF_W2  = OFF_W1 + 32 * 4;            // 32 f32
constexpr size_t OFF_RED = OFF_W2 + 32 * 4;            // 256 f32
constexpr size_t OFF_S1  = OFF_RED + 256 * 4;          // 64 f32
constexpr size_t SMEM_BYTES = OFF_S1 + 64 * 4;         // ~157.5 KB (< 320 KB/WGP)

typedef __bf16 bf16;
typedef __attribute__((ext_vector_type(16))) bf16  v16bf;
typedef __attribute__((ext_vector_type(8)))  float v8f;

union FragAB { v16bf v; uint4 q[2]; };
union FragD  { v8f v; float4 f[2]; };

__device__ __forceinline__ float sigf(float x) { return 1.0f / (1.0f + expf(-x)); }

// sense-reversing grid barrier, agent scope, fixed 16 workgroups
__device__ __forceinline__ void gbar(int* cnt, int* gen) {
  __syncthreads();
  if (threadIdx.x == 0) {
    __threadfence();
    int g = __hip_atomic_load(gen, __ATOMIC_ACQUIRE, __HIP_MEMORY_SCOPE_AGENT);
    int prev = __hip_atomic_fetch_add(cnt, 1, __ATOMIC_ACQ_REL, __HIP_MEMORY_SCOPE_AGENT);
    if (prev == kNWG - 1) {
      __hip_atomic_store(cnt, 0, __ATOMIC_RELAXED, __HIP_MEMORY_SCOPE_AGENT);
      __hip_atomic_fetch_add(gen, 1, __ATOMIC_ACQ_REL, __HIP_MEMORY_SCOPE_AGENT);
    } else {
      while (__hip_atomic_load(gen, __ATOMIC_ACQUIRE, __HIP_MEMORY_SCOPE_AGENT) == g) {
        __builtin_amdgcn_s_sleep(2);
      }
    }
  }
  __syncthreads();
}

__global__ void bilstm_init_kernel(int* bar) {
  if (threadIdx.x < 2) bar[threadIdx.x] = 0;
}

__global__ __launch_bounds__(kTPB)
void bilstm_attn_kernel(const float* __restrict__ x,
                        const float* __restrict__ Wih_f, const float* __restrict__ Whh_f,
                        const float* __restrict__ b_f,
                        const float* __restrict__ Wih_b, const float* __restrict__ Whh_b,
                        const float* __restrict__ b_b,
                        const float* __restrict__ wt,
                        float* __restrict__ out,
                        float* __restrict__ h_old,            // [B][TMAX][H2] f32
                        unsigned short* __restrict__ hbuf_u,  // [2 parity][2 dir][64 b][256 k] bf16
                        float* __restrict__ partS,            // [2 parity][16 wg][64 b]
                        float* __restrict__ partT,            // [16 wg][64 b]  (s1, early steps)
                        int* __restrict__ bar) {
  extern __shared__ char smem[];
  bf16*  Wl  = (bf16*) (smem + OFF_W);
  bf16*  Hl  = (bf16*) (smem + OFF_HL);
  float* Gl  = (float*)(smem + OFF_G);
  float* Cl  = (float*)(smem + OFF_C);
  float* Hs  = (float*)(smem + OFF_HS);
  float* Tv  = (float*)(smem + OFF_TV);
  int*   Ti  = (int*)  (smem + OFF_TI);
  float* Av  = (float*)(smem + OFF_AV);
  int*   Ai  = (int*)  (smem + OFF_AI);
  float* Xv  = (float*)(smem + OFF_XV);
  float* WihS= (float*)(smem + OFF_WIH);
  float* Bia = (float*)(smem + OFF_BIA);
  float* W1s = (float*)(smem + OFF_W1);
  float* W2s = (float*)(smem + OFF_W2);
  float* Red = (float*)(smem + OFF_RED);
  float* S1v = (float*)(smem + OFF_S1);
  bf16* hbuf = (bf16*)hbuf_u;

  const int tid  = threadIdx.x;
  const int lane = tid & 31;
  const int wv   = tid >> 5;          // wave id 0..7 == N-tile
  const int wg   = blockIdx.x;        // 0..15
  const int dir  = wg >> 3;           // 0 fwd, 1 bwd
  const int sub  = wg & 7;
  const int ch   = sub * 32;          // column base within direction
  const int hc0  = wg * 32;           // column base within h_comb

  const float* Whh = dir ? Whh_b : Whh_f;
  const float* Wih = dir ? Wih_b : Wih_f;
  const float* bia = dir ? b_b   : b_f;
  int* cnt = bar; int* gen = bar + 1;

  // ---------- preload: weight slice to LDS (bf16), bias/Wih/w1/w2 slices, state init ----------
  for (int idx = tid; idx < 128 * kH; idx += kTPB) {
    int n = idx >> 8, k = idx & 255;              // n = 32*q + u
    int gq = (n >> 5) * kH + ch + (n & 31);       // gate row in [0,4H)
    Wl[n * WST + k] = (bf16)Whh[gq * kH + k];
  }
  if (tid < 128) {
    int n = tid;
    int gq = (n >> 5) * kH + ch + (n & 31);
    WihS[n] = Wih[gq];
    Bia[n]  = bia[gq];
  }
  if (tid < 32) { W1s[tid] = wt[hc0 + tid]; W2s[tid] = wt[kH2 + hc0 + tid]; }
  for (int idx = tid; idx < 32 * 64; idx += kTPB) Cl[idx] = 0.0f;
  for (int idx = tid; idx < 5 * 64; idx += kTPB) { Tv[idx] = kNEG; Ti[idx] = 0; Av[idx] = 0.0f; Ai[idx] = 0; }
  // zero h state buffers (both parities) for our column slice, and h_old row 0 slice
  for (int idx = tid; idx < 2 * 32 * 64; idx += kTPB) {
    int par = idx >> 11, u = (idx >> 6) & 31, b = idx & 63;
    hbuf[(((par * 2 + dir) * 64 + b) * 256) + ch + u] = (bf16)0.0f;
  }
  for (int idx = tid; idx < 32 * 64; idx += kTPB) {
    int u = idx >> 6, b = idx & 63;
    h_old[((size_t)b * kTMAX) * kH2 + hc0 + u] = 0.0f;
  }
  gbar(cnt, gen);

  const int bb = tid & 63;        // batch for elementwise phases
  const int ug = tid >> 6;        // 0..3, owns u = ug*8 .. ug*8+7

  // ================= recurrence =================
  for (int i = 0; i < kS; ++i) {
    const int p = i & 1;

    // ---- phase 1: async-stage h_prev into LDS via the copy engine (ASYNCcnt path) ----
    // hbuf rows are [b][k]: each lane moves 16B chunks contiguous in global AND in LDS.
    {
      const bf16* gsrc = hbuf + ((size_t)(p * 2 + dir) * 64 + bb) * 256;
      const uint64_t gbase = (uint64_t)(uintptr_t)gsrc;
      const unsigned lbase = (unsigned)OFF_HL + (unsigned)bb * (WST * 2);
      #pragma unroll
      for (int c = 0; c < 8; ++c) {
        int ck = ug * 8 + c;                  // 16B chunk 0..31 within the 512B row
        unsigned lds_off = lbase + ck * 16;
        uint64_t ga = gbase + (uint64_t)ck * 16;
        asm volatile("global_load_async_to_lds_b128 %0, %1, off"
                     :: "v"(lds_off), "v"(ga) : "memory");
      }
    }
    if (tid < 64) Xv[tid] = x[tid * kS + (dir ? (kS - 1 - i) : i)];

    // ---- phase 2: insert score[t=i] into running top-5 (overlaps the async copy) ----
    if (tid < 64) {
      float val = 0.0f;
      if (i > 0) {
        const float* ps = partS + (size_t)((i - 1) & 1) * kNWG * 64;
        #pragma unroll
        for (int g = 0; g < kNWG; ++g) val += ps[g * 64 + tid];
      }
      if (val > Tv[4 * 64 + tid]) {
        int e = 4;
        while (e > 0 && val > Tv[(e - 1) * 64 + tid]) {
          Tv[e * 64 + tid] = Tv[(e - 1) * 64 + tid];
          Ti[e * 64 + tid] = Ti[(e - 1) * 64 + tid];
          --e;
        }
        Tv[e * 64 + tid] = val; Ti[e * 64 + tid] = i;
      }
    }
    asm volatile("s_wait_asynccnt 0" ::: "memory");
    __syncthreads();

    // ---- phase 3: GEMM  G[64 x 128] = h[64 x 256] * Wslice^T  (bf16 WMMA, f32 acc) ----
    {
      const int nt = wv;
      const int l16 = lane & 15;
      const int hi  = lane >> 4;
      v8f acc0 = {0,0,0,0,0,0,0,0}, acc1 = acc0, acc2 = acc0, acc3 = acc0;
      #pragma unroll
      for (int ks = 0; ks < 8; ++ks) {
        FragAB bfr;
        const uint4* wp = (const uint4*)(Wl + (nt * 16 + l16) * WST + ks * 32 + hi * 8);
        bfr.q[0] = wp[0]; bfr.q[1] = wp[2];
        #pragma unroll
        for (int mt = 0; mt < 4; ++mt) {
          FragAB afr;
          const bf16* ap = Hl + (mt * 16 + l16) * WST + ks * 32 + hi * 8;
          afr.q[0] = *(const uint4*)(ap);
          afr.q[1] = *(const uint4*)(ap + 16);
          v8f* acc = (mt == 0) ? &acc0 : (mt == 1) ? &acc1 : (mt == 2) ? &acc2 : &acc3;
          *acc = __builtin_amdgcn_wmma_f32_16x16x32_bf16(
              false, afr.v, false, bfr.v, (short)0, *acc, false, false);
        }
      }
      #pragma unroll
      for (int mt = 0; mt < 4; ++mt) {
        FragD d; d.v = (mt == 0) ? acc0 : (mt == 1) ? acc1 : (mt == 2) ? acc2 : acc3;
        float* gp = Gl + (nt * 16 + l16) * GST + mt * 16 + hi * 8;
        *(float4*)(gp)     = d.f[0];
        *(float4*)(gp + 4) = d.f[1];
      }
    }
    __syncthreads();

    // ---- phase 4: LSTM gates / cell update for our 32 columns ----
    #pragma unroll
    for (int j = 0; j < 8; ++j) {
      int u = ug * 8 + j;
      float xw = Xv[bb];
      float gi = Gl[(u)        * GST + bb] + Bia[u]      + xw * WihS[u];
      float gf = Gl[(32 + u)   * GST + bb] + Bia[32 + u] + xw * WihS[32 + u];
      float gg = Gl[(64 + u)   * GST + bb] + Bia[64 + u] + xw * WihS[64 + u];
      float go = Gl[(96 + u)   * GST + bb] + Bia[96 + u] + xw * WihS[96 + u];
      float c  = Cl[u * 64 + bb];
      float c2 = sigf(gf) * c + sigf(gi) * tanhf(gg);
      Cl[u * 64 + bb] = c2;
      float h = sigf(go) * tanhf(c2);
      Hs[u * 64 + bb] = h;
      hbuf[(((p ^ 1) * 2 + dir) * 64 + bb) * 256 + ch + u] = (bf16)h;
    }

    // ---- phase 5: attention weights ----
    const bool early = (i < kTOPK);   // rs = i+1 <= TOPK -> raw branch, needs s1
    if (early) {
      float ps = 0.0f;
      #pragma unroll
      for (int j = 0; j < 8; ++j) { int u = ug * 8 + j; ps += tanhf(Hs[u * 64 + bb]) * W1s[u]; }
      Red[tid] = ps;
      __syncthreads();
      if (tid < 64) partT[wg * 64 + tid] = Red[tid] + Red[64 + tid] + Red[128 + tid] + Red[192 + tid];
      gbar(cnt, gen);                 // cross-WG reduce of s1 (only 5 steps pay this)
      if (tid < 64) {
        float s1 = 0.0f;
        #pragma unroll
        for (int g = 0; g < kNWG; ++g) s1 += partT[g * 64 + tid];
        S1v[tid] = s1;
      }
      __syncthreads();
      if (tid < 64) {
        #pragma unroll
        for (int e = 0; e < 5; ++e) {
          if (e <= i) { Av[e * 64 + tid] = S1v[tid] + Tv[e * 64 + tid]; Ai[e * 64 + tid] = Ti[e * 64 + tid]; }
          else        { Av[e * 64 + tid] = 0.0f;                        Ai[e * 64 + tid] = 0; }
        }
      }
    } else {
      if (tid < 64) {
        float delta = Tv[4 * 64 + tid] + kEPS;   // s1 cancels in sparse branch
        float w[5]; float s = 0.0f;
        #pragma unroll
        for (int e = 0; e < 5; ++e) { w[e] = fmaxf(Tv[e * 64 + tid] - delta, 0.0f); s += w[e]; }
        float inv = 1.0f / (s + kEPS);
        #pragma unroll
        for (int e = 0; e < 5; ++e) { Av[e * 64 + tid] = w[e] * inv; Ai[e * 64 + tid] = Ti[e * 64 + tid]; }
      }
    }
    __syncthreads();

    // ---- phase 6: sparse attn_c gather (<=5 rows), h_new row store, score partial ----
    float ps2 = 0.0f;
    #pragma unroll
    for (int j = 0; j < 8; ++j) {
      int u = ug * 8 + j;
      float ac = 0.0f;
      #pragma unroll
      for (int e = 0; e < 5; ++e) {
        float w = Av[e * 64 + bb];
        int   t = Ai[e * 64 + bb];
        ac += w * h_old[((size_t)bb * kTMAX + t) * kH2 + hc0 + u];
      }
      float hn = Hs[u * 64 + bb] + ac;
      if (i == kS - 1) out[bb * kH2 + hc0 + u] = ac;      // output attn_c
      h_old[((size_t)bb * kTMAX + (i + 1)) * kH2 + hc0 + u] = hn;
      ps2 += tanhf(hn) * W2s[u];
    }
    Red[tid] = ps2;
    __syncthreads();
    if (tid < 64) partS[(size_t)p * kNWG * 64 + wg * 64 + tid] =
        Red[tid] + Red[64 + tid] + Red[128 + tid] + Red[192 + tid];

    // ---- final attn_w vector (zeros + <=5 scatter), WG0 only ----
    if (i == kS - 1 && wg == 0) {
      for (int idx = tid; idx < kB * kS; idx += kTPB) out[kB * kH2 + idx] = 0.0f;
      __threadfence_block();
      __syncthreads();
      if (tid < 64) {
        #pragma unroll
        for (int e = 0; e < 5; ++e) {
          float w = Av[e * 64 + tid];
          if (w != 0.0f) out[kB * kH2 + tid * kS + Ai[e * 64 + tid]] = w;
        }
      }
    }

    gbar(cnt, gen);
  }
}

extern "C" void kernel_launch(void* const* d_in, const int* in_sizes, int n_in,
                              void* d_out, int out_size, void* d_ws, size_t ws_size,
                              hipStream_t stream) {
  (void)in_sizes; (void)n_in; (void)out_size; (void)ws_size;
  const float* x     = (const float*)d_in[0];
  const float* Wih_f = (const float*)d_in[1];
  const float* Whh_f = (const float*)d_in[2];
  const float* b_f   = (const float*)d_in[3];
  const float* Wih_b = (const float*)d_in[4];
  const float* Whh_b = (const float*)d_in[5];
  const float* b_b   = (const float*)d_in[6];
  const float* wt    = (const float*)d_in[7];
  float* out = (float*)d_out;

  char* ws = (char*)d_ws;
  size_t off = 0;
  float* h_old = (float*)(ws + off); off += (size_t)kB * kTMAX * kH2 * sizeof(float);   // ~67.2 MB
  unsigned short* hbuf = (unsigned short*)(ws + off); off += (size_t)2 * 2 * 64 * 256 * 2;
  float* partS = (float*)(ws + off); off += (size_t)2 * kNWG * 64 * sizeof(float);
  float* partT = (float*)(ws + off); off += (size_t)kNWG * 64 * sizeof(float);
  int* bar = (int*)(ws + off); off += 2 * sizeof(int);

  bilstm_init_kernel<<<1, 64, 0, stream>>>(bar);
  bilstm_attn_kernel<<<kNWG, kTPB, SMEM_BYTES, stream>>>(
      x, Wih_f, Whh_f, b_f, Wih_b, Whh_b, b_b, wt, out,
      h_old, hbuf, partS, partT, bar);
}
